// CortexMQuantizableLSTM_39848706573661
// MI455X (gfx1250) — matrix operand order repaired and checked by reading, not compiled, not run on hardware
//
#include <hip/hip_runtime.h>
#include <stdint.h>
#include <stddef.h>

// ---------------- problem dims ----------------
#define T_STEPS 512
#define BATCH   64
#define IDIM    512
#define HDIM    512
#define KTOT    (IDIM + HDIM)   // 1024: A = [x_t | h]
#define NKC     (KTOT / 32)     // 32 k-chunks of 32
#define NKC_X   (IDIM / 32)     // 16 chunks from x
#define NKC_H   (HDIM / 32)     // 16 chunks from h

// ---------------- launch shape ----------------
#define NWG          32         // one WG per 16-wide h-column group
#define WAVES_PER_WG 4          // one wave per 16-row batch tile
#define THREADS      (WAVES_PER_WG * 32)

// global fragment buffers: lane-major [chunk][lane][j], 256 words / chunk
#define CHUNK_WORDS 256
#define HSW_WORDS (4 * NKC_H * CHUNK_WORDS)               // one h buffer: 64 KB
#define XSW_WORDS ((size_t)T_STEPS * 4 * NKC_X * CHUNK_WORDS) // x bf16: 33.5 MB
#define CTRL_PAD  16

// LDS B blocks: padded lane-major, lane base = lane*8 + (lane>>3)*4
#define BPAD_WORDS 272                                    // 32*8 + 16 pad
#define BLDS_WORDS (4 * NKC * BPAD_WORDS)                 // 139 KB

typedef __attribute__((ext_vector_type(16))) __bf16 v16bf;
typedef __attribute__((ext_vector_type(8)))  float  v8f;

union frag8 { uint32_t u[8]; uint4 q[2]; v16bf v; };

__device__ __forceinline__ uint16_t f2bf(float f) {
  union { float f; uint32_t u; } v; v.f = f;
  uint32_t r = v.u + 0x7FFFu + ((v.u >> 16) & 1u);   // round-to-nearest-even
  return (uint16_t)(r >> 16);
}

__device__ __forceinline__ uint32_t pack_bf16(float a, float b) {
#if __has_builtin(__builtin_amdgcn_cvt_pk_bf16_f32)
  auto r = __builtin_amdgcn_cvt_pk_bf16_f32(a, b);   // v_cvt_pk_bf16_f32
  uint32_t u; __builtin_memcpy(&u, &r, 4);
  return u;
#else
  return (uint32_t)f2bf(a) | ((uint32_t)f2bf(b) << 16);
#endif
}

__device__ __forceinline__ float fast_tanh(float x) {
#if __has_builtin(__builtin_amdgcn_tanhf)
  return __builtin_amdgcn_tanhf(x);                  // v_tanh_f32
#elif __has_builtin(__builtin_amdgcn_tanh_f32)
  return __builtin_amdgcn_tanh_f32(x);
#else
  return tanhf(x);
#endif
}

__device__ __forceinline__ float sigm(float x) {
  return 0.5f * (1.0f + fast_tanh(0.5f * x));        // 1 trans op
}

// zero both h ping-pong buffers + barrier counter (graph-replay safe re-init)
__global__ void lstm_init_kernel(uint32_t* ws, int nwords) {
  for (int i = threadIdx.x; i < nwords; i += blockDim.x) ws[i] = 0u;
}

// one-time parallel pre-pass: x (fp32) -> bf16 pairs, WMMA-A fragments,
// lane-major: xsw[(t*4+r)*NKC_X + kc][lane][j]
__global__ void lstm_xswizzle_kernel(const float* __restrict__ x,
                                     uint32_t* __restrict__ xsw) {
  const size_t npairs = (size_t)T_STEPS * BATCH * (IDIM / 2);
  for (size_t p = (size_t)blockIdx.x * blockDim.x + threadIdx.x;
       p < npairs; p += (size_t)gridDim.x * blockDim.x) {
    int    kp = (int)(p & (IDIM / 2 - 1));       // 0..255
    int    m  = (int)((p >> 8) & (BATCH - 1));   // 0..63
    size_t t  = p >> 14;
    int k  = 2 * kp;
    int kc = k >> 5;
    int ko = k & 31;
    int j  = ((ko >> 4) & 1) * 4 + ((ko >> 1) & 3);
    int hi = (ko >> 3) & 1;
    int la = hi * 16 + (m & 15);
    int r  = m >> 4;
    float2 w = *(const float2*)(x + ((size_t)t * BATCH + m) * IDIM + k);
    size_t dst = ((t * 4 + r) * NKC_X + kc) * CHUNK_WORDS + (size_t)la * 8 + j;
    xsw[dst] = pack_bf16(w.x, w.y);
  }
}

template <bool XPRE>
__global__ __launch_bounds__(THREADS)
void lstm_persistent_kernel(const float* __restrict__ x,     // [T][B][I]
                            const float* __restrict__ Wih,   // [4H][I]
                            const float* __restrict__ Whh,   // [4H][H]
                            const float* __restrict__ bih,   // [4H]
                            const float* __restrict__ bhh,   // [4H]
                            float*       __restrict__ out,   // [T][B][H]
                            uint32_t*    __restrict__ hsw,   // 2 ping-pong h buffers
                            const uint32_t* __restrict__ xsw,// swizzled bf16 x (XPRE)
                            uint32_t*    __restrict__ ctr)   // cumulative arrive counter
{
  // B operand, pre-swizzled bf16, per-WG slice of [Wih;Whh]^T.
  // Padded lane-major blocks -> 2x ds_load_b128 per fragment, 2-way conflict.
  __shared__ __attribute__((aligned(16))) uint32_t Blds[BLDS_WORDS];

  const int cg   = blockIdx.x;          // h-column group: cols 16*cg..16*cg+15
  const int lane = threadIdx.x & 31;
  const int r    = threadIdx.x >> 5;    // batch row tile: rows 16*r..16*r+15
  const int n    = lane & 15;           // column within 16-wide tile
  const int hi   = lane >> 4;           // lane half
  const int m    = r * 16 + n;          // batch row this lane's A-fragment serves
  const int lofs = lane * 8 + (lane >> 3) * 4;   // padded LDS lane base

  // ---- one-time: stage weights into LDS in exact WMMA-B layout ----
  // B(32x16 bf16) chunk: VGPR j of lane half kh holds K = 16*(j>>2) + 8*kh + 2*(j&3) + {0,1};
  // lane%16 = output column.
  for (int e = threadIdx.x; e < 4 * NKC * 8 * 32; e += THREADS) {
    int ls = e & 31;
    int j  = (e >> 5) & 7;
    int kc = (e >> 8) & 31;
    int g  = (e >> 13) & 3;
    int col  = ls & 15;
    int kh   = ls >> 4;
    int K    = kc * 32 + (j >> 2) * 16 + kh * 8 + 2 * (j & 3);     // even, never straddles I/H split
    int gcol = g * HDIM + cg * 16 + col;
    float w0, w1;
    if (K < IDIM) { w0 = Wih[(size_t)gcol * IDIM + K];
                    w1 = Wih[(size_t)gcol * IDIM + K + 1]; }
    else          { w0 = Whh[(size_t)gcol * HDIM + (K - IDIM)];
                    w1 = Whh[(size_t)gcol * HDIM + (K - IDIM) + 1]; }
    Blds[(size_t)(g * NKC + kc) * BPAD_WORDS + (ls * 8 + (ls >> 3) * 4) + j]
        = pack_bf16(w0, w1);
  }
  __syncthreads();

  // per-lane gate biases (column-constant across the C/D tile)
  float bias[4];
  #pragma unroll
  for (int g = 0; g < 4; ++g) {
    int gc = g * HDIM + cg * 16 + n;
    bias[g] = bih[gc] + bhh[gc];
  }

  // precomputed h-store swizzle for this lane's column (k fixed per lane)
  const int kq    = cg * 16 + n;
  const int kcx   = kq >> 5;
  const int kp    = kq & 31;
  const int jj    = ((kp >> 4) & 1) * 4 + ((kp >> 1) & 3);
  const int hh    = (kp >> 3) & 1;
  const int halfb = kp & 1;

  v8f cst = {};   // cell state lives in registers for the whole sequence

  for (int t = 0; t < T_STEPS; ++t) {
    v8f acc[4];
    #pragma unroll
    for (int g = 0; g < 4; ++g) {
      v8f a;
      #pragma unroll
      for (int v = 0; v < 8; ++v) a[v] = bias[g];
      acc[g] = a;
    }

    // ================= phase 1: x-projection (independent of h) ==========
    if (t + 1 < T_STEPS) {   // warm L2 for next timestep's x slice
      if (XPRE) {
        const char* nx = (const char*)(xsw + ((size_t)((t + 1) * 4 + r) * NKC_X) * CHUNK_WORDS);
        #pragma unroll
        for (int q = 0; q < 4; ++q)
          __builtin_prefetch(nx + q * 4096 + lane * 128, 0, 0);
      } else {
        const char* nx = (const char*)(x + ((size_t)(t + 1) * BATCH + m) * IDIM);
        #pragma unroll 4
        for (int kc = 0; kc < NKC_X; ++kc)
          __builtin_prefetch(nx + kc * 128, 0, 0);
      }
    }

    const uint32_t* xsl  = xsw + ((size_t)(t * 4 + r) * NKC_X) * CHUNK_WORDS + lane * 8;
    const float*    xrow = x + ((size_t)t * BATCH + m) * IDIM;
    #pragma unroll 2
    for (int kc = 0; kc < NKC_X; ++kc) {
      frag8 A;
      if (XPRE) {
        const uint4* xp = (const uint4*)(xsl + (size_t)kc * CHUNK_WORDS);
        A.q[0] = xp[0];
        A.q[1] = xp[1];
      } else {
        const float2* xb = (const float2*)(xrow + kc * 32);
        #pragma unroll
        for (int j = 0; j < 8; ++j) {
          int k0 = (j < 4 ? 2 * j : 16 + 2 * (j - 4)) + (hi ? 8 : 0);
          float2 w = xb[k0 >> 1];
          A.u[j] = pack_bf16(w.x, w.y);
        }
      }
      #pragma unroll
      for (int g = 0; g < 4; ++g) {
        frag8 Bm;
        const uint4* bp = (const uint4*)(&Blds[(size_t)(g * NKC + kc) * BPAD_WORDS + lofs]);
        Bm.q[0] = bp[0];
        Bm.q[1] = bp[1];
        acc[g] = __builtin_amdgcn_wmma_f32_16x16x32_bf16(
            false, A.v, false, Bm.v, (short)0, acc[g], false, false);
      }
    }

    // ================= barrier WAIT: h(t-1) must be fully published =======
    if (threadIdx.x == 0) {
      unsigned tgt = (unsigned)(NWG * t);
      while (__hip_atomic_load(ctr, __ATOMIC_ACQUIRE, __HIP_MEMORY_SCOPE_AGENT) < tgt) {
        __builtin_amdgcn_s_sleep(2);
      }
    }
    __syncthreads();

    // ================= phase 2: recurrent h-projection ====================
    const uint32_t* hbuf = hsw + (size_t)(t & 1) * HSW_WORDS + (size_t)r * NKC_H * CHUNK_WORDS + lane * 8;
    #pragma unroll 2
    for (int kc = 0; kc < NKC_H; ++kc) {
      frag8 A;
      const uint4* hp = (const uint4*)(hbuf + (size_t)kc * CHUNK_WORDS);
      A.q[0] = hp[0];
      A.q[1] = hp[1];
      #pragma unroll
      for (int g = 0; g < 4; ++g) {
        frag8 Bm;
        const uint4* bp = (const uint4*)(&Blds[(size_t)(g * NKC + (NKC_X + kc)) * BPAD_WORDS + lofs]);
        Bm.q[0] = bp[0];
        Bm.q[1] = bp[1];
        acc[g] = __builtin_amdgcn_wmma_f32_16x16x32_bf16(
            false, A.v, false, Bm.v, (short)0, acc[g], false, false);
      }
    }

    // ================= phase 3: cell update + publish h(t) ================
    uint16_t* wbuf16 = (uint16_t*)(hsw + (size_t)((t + 1) & 1) * HSW_WORDS);
    #pragma unroll
    for (int v = 0; v < 8; ++v) {
      float ig = sigm(acc[0][v]);
      float fg = sigm(acc[1][v]);
      float gg = fast_tanh(acc[2][v]);
      float og = sigm(acc[3][v]);
      float cv = fg * cst[v] + ig * gg;
      cst[v] = cv;
      float hv = og * fast_tanh(cv);

      int mm   = v + 8 * hi;              // row within tile (C/D layout)
      int mrow = r * 16 + mm;             // batch row
      out[((size_t)t * BATCH + mrow) * HDIM + cg * 16 + n] = hv;

      // scatter h into lane-major WMMA-A fragment layout for next step
      int la = hh * 16 + mm;
      uint32_t idx = (uint32_t)(r * NKC_H + kcx) * CHUNK_WORDS + la * 8 + jj;
      wbuf16[idx * 2 + halfb] = f2bf(hv);
    }

    // ================= barrier ARRIVE: h(t) published =====================
    __threadfence();
    __syncthreads();
    if (threadIdx.x == 0) {
      __hip_atomic_fetch_add(ctr, 1u, __ATOMIC_ACQ_REL, __HIP_MEMORY_SCOPE_AGENT);
    }
    __syncthreads();
  }
}

extern "C" void kernel_launch(void* const* d_in, const int* in_sizes, int n_in,
                              void* d_out, int out_size, void* d_ws, size_t ws_size,
                              hipStream_t stream) {
  const float* x   = (const float*)d_in[0];
  const float* Wih = (const float*)d_in[1];
  const float* Whh = (const float*)d_in[2];
  const float* bih = (const float*)d_in[3];
  const float* bhh = (const float*)d_in[4];

  uint32_t* hsw = (uint32_t*)d_ws;
  uint32_t* ctr = hsw + 2 * HSW_WORDS;
  uint32_t* xsw = hsw + 2 * HSW_WORDS + CTRL_PAD;

  const size_t need = (2 * (size_t)HSW_WORDS + CTRL_PAD + XSW_WORDS) * 4;
  const bool xpre = (ws_size >= need);

  lstm_init_kernel<<<1, 256, 0, stream>>>(hsw, 2 * HSW_WORDS + 1);
  if (xpre) {
    lstm_xswizzle_kernel<<<2048, 256, 0, stream>>>(x, xsw);
    lstm_persistent_kernel<true><<<NWG, THREADS, 0, stream>>>(
        x, Wih, Whh, bih, bhh, (float*)d_out, hsw, xsw, ctr);
  } else {
    lstm_persistent_kernel<false><<<NWG, THREADS, 0, stream>>>(
        x, Wih, Whh, bih, bhh, (float*)d_out, hsw, xsw, ctr);
  }
}
